// ConditionalGraphNetwork_49246095016470
// MI455X (gfx1250) — matrix operand order repaired and checked by reading, not compiled, not run on hardware
//
#include <hip/hip_runtime.h>

// ============================================================================
// ConditionalGraphNetwork (MeshGraphNets-style) for MI455X / gfx1250.
//
// Strategy (see analysis): all 64-wide MLP matmuls run on
// v_wmma_f32_16x16x32_bf16 (bf16 inputs, f32 accumulate). Weights are packed
// once per call into transposed bf16 "Wt[n][k]" so B fragments are two
// contiguous 16B loads. Activations staged to LDS as bf16 per 64-row
// workgroup; each wave (32 lanes) owns 16 rows and computes a 16x64 tile with
// 4 accumulators. Scatter-mean via f32 global atomics + precomputed 1/deg.
// Working set (~150MB) is L2-resident on the 192MB L2.
// ============================================================================

#define NODE_IN 16
#define EDGE_IN 4
#define COND_IN 10
#define NOUT    3
#define NB      8
#define H       64
#define NLAYERS 3

typedef __attribute__((ext_vector_type(16))) __bf16 bf16x16;
typedef __attribute__((ext_vector_type(8)))  float  f32x8;

union ABFrag { bf16x16 v; uint4 q[2]; };

__device__ __forceinline__ void atomAddF(float* p, float v) {
  (void)__hip_atomic_fetch_add(p, v, __ATOMIC_RELAXED, __HIP_MEMORY_SCOPE_AGENT);
}

// ----------------------------------------------------------------------------
// Wave-level GEMM tile: C[16x64] (f32) += A[16xK] (bf16, LDS) * W[Kx64] (bf16).
// A rows have stride ASTR bf16 elements; W is packed transposed: Wt[n][k],
// row stride BSTR. Fragment layouts per CDNA5 ISA (wave32):
//   A elem e: K = 16*(e>=8) + 8*half + (e&7)   -> two contiguous 16B loads
//   B elem e: K = 16*(e>=8) + 8*half + (e&7)   -> two contiguous 16B loads
//   C: col = lane&15 (+16*t), row = r + 8*half
// ----------------------------------------------------------------------------
template<int KSTEPS, int ASTR, int BSTR>
__device__ __forceinline__ void wave_gemm(const __bf16* A0, const __bf16* Wt,
                                          f32x8 acc[4], int l15, int hi) {
#pragma unroll
  for (int s = 0; s < KSTEPS; ++s) {
    const int k0 = s * 32;
    ABFrag a;
    const __bf16* ap = A0 + l15 * ASTR + k0 + hi * 8;
    a.q[0] = *(const uint4*)(ap);
    a.q[1] = *(const uint4*)(ap + 16);
#pragma unroll
    for (int t = 0; t < 4; ++t) {
      ABFrag b;
      const __bf16* bp = Wt + (t * 16 + l15) * BSTR + k0 + hi * 8;
      b.q[0] = *(const uint4*)(bp);
      b.q[1] = *(const uint4*)(bp + 16);
      acc[t] = __builtin_amdgcn_wmma_f32_16x16x32_bf16(
          false, a.v, false, b.v, (short)0, acc[t], false, false);
    }
  }
}

// bias + relu + store hidden tile to this wave's 16x64 bf16 LDS slab
__device__ __forceinline__ void store_hidden_relu(__bf16* Hw, const f32x8 acc[4],
                                                  const float* bias, int l15, int hi) {
#pragma unroll
  for (int t = 0; t < 4; ++t) {
    float bn = bias[t * 16 + l15];
#pragma unroll
    for (int r = 0; r < 8; ++r) {
      float v = fmaxf(acc[t][r] + bn, 0.0f);
      Hw[(hi * 8 + r) * H + t * 16 + l15] = (__bf16)v;
    }
  }
}

// ------------------------- staging helpers (f32 -> bf16) --------------------
__device__ __forceinline__ void cvt4(__bf16* d, float4 f) {
  union { __bf16 h[4]; uint2 u; } p;
  p.h[0] = (__bf16)f.x; p.h[1] = (__bf16)f.y;
  p.h[2] = (__bf16)f.z; p.h[3] = (__bf16)f.w;
  *(uint2*)d = p.u;
}
__device__ __forceinline__ void stage32(__bf16* d, const float* s) {
  const float4* s4 = (const float4*)s;
#pragma unroll
  for (int j = 0; j < 8; ++j) cvt4(d + 4 * j, s4[j]);
}
__device__ __forceinline__ void stage32_scaled(__bf16* d, const float* s, float sc) {
  const float4* s4 = (const float4*)s;
#pragma unroll
  for (int j = 0; j < 8; ++j) {
    float4 f = s4[j];
    f.x *= sc; f.y *= sc; f.z *= sc; f.w *= sc;
    cvt4(d + 4 * j, f);
  }
}
__device__ __forceinline__ void zero16(__bf16* d) {
  uint4 z = {0, 0, 0, 0};
  ((uint4*)d)[0] = z; ((uint4*)d)[1] = z;
}

// ============================ small utility kernels =========================
__global__ void k_zero(float* p, int n) {
  for (int i = blockIdx.x * blockDim.x + threadIdx.x; i < n;
       i += gridDim.x * blockDim.x) p[i] = 0.0f;
}

// pack w[din,64] (f32 row-major) -> Wt[64][kp] bf16 transposed, zero-padded
__global__ void k_pack_w(const float* __restrict__ w, __bf16* __restrict__ dst,
                         int din, int kp) {
  int i = blockIdx.x * blockDim.x + threadIdx.x;
  if (i >= 64 * kp) return;
  int n = i / kp, k = i % kp;
  float v = (k < din) ? w[k * 64 + n] : 0.0f;
  dst[i] = (__bf16)v;
}

__global__ void k_deg_count(const int* __restrict__ col, float* __restrict__ deg, int Ee) {
  int i = blockIdx.x * blockDim.x + threadIdx.x;
  if (i < Ee) atomAddF(&deg[col[i]], 1.0f);
}
__global__ void k_deg_inv(float* deg, int n) {
  int i = blockIdx.x * blockDim.x + threadIdx.x;
  if (i < n) deg[i] = 1.0f / fmaxf(deg[i], 1.0f);
}

// cond encoder: tiny (8x10 -> 8x64), plain f32
__global__ void k_cond_enc(const float* __restrict__ cond,
                           const float* __restrict__ w1, const float* __restrict__ b1,
                           const float* __restrict__ w2, const float* __restrict__ b2,
                           float* __restrict__ u) {
  __shared__ float h[H];
  int n = threadIdx.x;  // 64 threads
  for (int b = 0; b < NB; ++b) {
    float s = b1[n];
    for (int k = 0; k < COND_IN; ++k) s += cond[b * COND_IN + k] * w1[k * H + n];
    h[n] = fmaxf(s, 0.0f);
    __syncthreads();
    float o = b2[n];
    for (int k = 0; k < H; ++k) o += h[k] * w2[k * H + n];
    u[b * H + n] = o;
    __syncthreads();
  }
}

// ============================ WMMA MLP kernels ==============================
// Node encoder: x[N,16] -> relu -> [N,64]. K padded 16->32.
__global__ void __launch_bounds__(128) k_node_enc(
    const float* __restrict__ xin,
    const __bf16* __restrict__ W1t, const float* __restrict__ b1,
    const __bf16* __restrict__ W2t, const float* __restrict__ b2,
    float* __restrict__ xout, int Nn) {
  __shared__ __attribute__((aligned(16))) __bf16 A[64 * 32];
  __shared__ __attribute__((aligned(16))) __bf16 Hd[64 * H];
  const int wg0 = blockIdx.x * 64;
  const int tid = threadIdx.x;
  {
    const int r = tid >> 1, hf = tid & 1;
    const int g = wg0 + r;
    __bf16* dst = A + r * 32 + hf * 16;
    zero16(dst);
    if (hf == 0 && g < Nn) {
      const float4* s4 = (const float4*)(xin + g * NODE_IN);
#pragma unroll
      for (int j = 0; j < 4; ++j) cvt4(dst + 4 * j, s4[j]);
    }
  }
  const int wave = tid >> 5, lane = tid & 31, l15 = lane & 15, hi = lane >> 4;
  f32x8 acc[4] = {};
  wave_gemm<1, 32, 32>(A + wave * 16 * 32, W1t, acc, l15, hi);
  __bf16* Hw = Hd + wave * 16 * H;
  store_hidden_relu(Hw, acc, b1, l15, hi);
  f32x8 acc2[4] = {};
  wave_gemm<2, H, H>(Hw, W2t, acc2, l15, hi);
  const int r0 = wg0 + wave * 16 + hi * 8;
#pragma unroll
  for (int t = 0; t < 4; ++t) {
    float bn = b2[t * 16 + l15];
#pragma unroll
    for (int r = 0; r < 8; ++r) {
      int g = r0 + r;
      if (g < Nn) xout[g * H + t * 16 + l15] = acc2[t][r] + bn;
    }
  }
}

// Edge encoder: edge_attr[E,4] -> [E,64]. K padded 4->32.
__global__ void __launch_bounds__(128) k_edge_enc(
    const float* __restrict__ eattr,
    const __bf16* __restrict__ W1t, const float* __restrict__ b1,
    const __bf16* __restrict__ W2t, const float* __restrict__ b2,
    float* __restrict__ eaout, int Ee) {
  __shared__ __attribute__((aligned(16))) __bf16 A[64 * 32];
  __shared__ __attribute__((aligned(16))) __bf16 Hd[64 * H];
  const int wg0 = blockIdx.x * 64;
  const int tid = threadIdx.x;
  {
    const int r = tid >> 1, hf = tid & 1;
    const int g = wg0 + r;
    __bf16* dst = A + r * 32 + hf * 16;
    zero16(dst);
    if (hf == 0 && g < Ee) {
      float4 f = *(const float4*)(eattr + g * EDGE_IN);
      cvt4(dst, f);
    }
  }
  const int wave = tid >> 5, lane = tid & 31, l15 = lane & 15, hi = lane >> 4;
  f32x8 acc[4] = {};
  wave_gemm<1, 32, 32>(A + wave * 16 * 32, W1t, acc, l15, hi);
  __bf16* Hw = Hd + wave * 16 * H;
  store_hidden_relu(Hw, acc, b1, l15, hi);
  f32x8 acc2[4] = {};
  wave_gemm<2, H, H>(Hw, W2t, acc2, l15, hi);
  const int r0 = wg0 + wave * 16 + hi * 8;
#pragma unroll
  for (int t = 0; t < 4; ++t) {
    float bn = b2[t * 16 + l15];
#pragma unroll
    for (int r = 0; r < 8; ++r) {
      int g = r0 + r;
      if (g < Ee) eaout[g * H + t * 16 + l15] = acc2[t][r] + bn;
    }
  }
}

// Fused edge model + node1 message model + scatter-add.
//   e_in = [x[row] | x[col] | ea | u[batch[row]]]  (K=256) -> ea_new
//   msg  = MLP([x[row] | ea_new], K=128)           -> atomicAdd agg[col]
__global__ void __launch_bounds__(128) k_edge_layer(
    const float* __restrict__ x, const int* __restrict__ ei,
    const float* __restrict__ u, const int* __restrict__ batch,
    float* __restrict__ ea,
    const __bf16* __restrict__ We1t, const float* __restrict__ be1,
    const __bf16* __restrict__ We2t, const float* __restrict__ be2,
    const __bf16* __restrict__ Wn1t, const float* __restrict__ bn1,
    const __bf16* __restrict__ Wn2t, const float* __restrict__ bn2,
    float* __restrict__ agg, int Ee) {
  __shared__ __attribute__((aligned(16))) __bf16 A[64 * 256];
  __shared__ __attribute__((aligned(16))) __bf16 Hd[64 * H];
  const int wg0 = blockIdx.x * 64;
  const int tid = threadIdx.x;
  const int* rowI = ei;
  const int* colI = ei + Ee;
  {
    const int e = tid >> 1, hf = tid & 1;
    const int ge = wg0 + e;
    __bf16* dst = A + e * 256 + hf * 32;   // covers [seg*64 + hf*32] via offsets
    if (ge < Ee) {
      const int rs = rowI[ge];
      const int cd = colI[ge];
      const int bb = batch[rs];
      stage32(dst + 0,   x  + rs * H + hf * 32);
      stage32(dst + 64,  x  + cd * H + hf * 32);
      stage32(dst + 128, ea + (size_t)ge * H + hf * 32);
      stage32(dst + 192, u  + bb * H + hf * 32);
    } else {
      uint4 z = {0, 0, 0, 0};
#pragma unroll
      for (int sg = 0; sg < 4; ++sg)
#pragma unroll
        for (int j = 0; j < 4; ++j) ((uint4*)(dst + sg * 64))[j] = z;
    }
  }
  const int wave = tid >> 5, lane = tid & 31, l15 = lane & 15, hi = lane >> 4;
  const __bf16* Aw = A + wave * 16 * 256;
  __bf16* Hw = Hd + wave * 16 * H;

  // ---- edge MLP: K=256 -> 64 -> 64 ----
  f32x8 acc[4] = {};
  wave_gemm<8, 256, 256>(Aw, We1t, acc, l15, hi);
  store_hidden_relu(Hw, acc, be1, l15, hi);
  f32x8 accE[4] = {};
  wave_gemm<2, H, H>(Hw, We2t, accE, l15, hi);

  const int er0 = wg0 + wave * 16 + hi * 8;
#pragma unroll
  for (int t = 0; t < 4; ++t) {
    float bn = be2[t * 16 + l15];
#pragma unroll
    for (int r = 0; r < 8; ++r) {
      float v = accE[t][r] + bn;
      int lrow = wave * 16 + hi * 8 + r;
      A[lrow * 256 + 64 + t * 16 + l15] = (__bf16)v;  // node1 A = [x[row]|ea_new]
      int ge = er0 + r;
      if (ge < Ee) ea[(size_t)ge * H + t * 16 + l15] = v;
    }
  }

  // ---- node1 message MLP: K=128 -> 64 -> 64 ----
  f32x8 accH[4] = {};
  wave_gemm<4, 256, 128>(Aw, Wn1t, accH, l15, hi);
  store_hidden_relu(Hw, accH, bn1, l15, hi);
  f32x8 accM[4] = {};
  wave_gemm<2, H, H>(Hw, Wn2t, accM, l15, hi);

  int cols[8];
#pragma unroll
  for (int r = 0; r < 8; ++r) {
    int ge = er0 + r;
    cols[r] = (ge < Ee) ? colI[ge] : -1;
  }
#pragma unroll
  for (int t = 0; t < 4; ++t) {
    float bn = bn2[t * 16 + l15];
#pragma unroll
    for (int r = 0; r < 8; ++r)
      if (cols[r] >= 0)
        atomAddF(&agg[(size_t)cols[r] * H + t * 16 + l15], accM[t][r] + bn);
  }
}

// Node update: x' = MLP([x | agg/deg | u[batch]], K=192) + x  (residual)
__global__ void __launch_bounds__(128) k_node_layer(
    const float* __restrict__ x, const float* __restrict__ agg,
    const float* __restrict__ dinv, const float* __restrict__ u,
    const int* __restrict__ batch,
    const __bf16* __restrict__ W1t, const float* __restrict__ b1,
    const __bf16* __restrict__ W2t, const float* __restrict__ b2,
    float* __restrict__ xout, int Nn) {
  __shared__ __attribute__((aligned(16))) __bf16 A[64 * 192];
  __shared__ __attribute__((aligned(16))) __bf16 Hd[64 * H];
  const int wg0 = blockIdx.x * 64;
  const int tid = threadIdx.x;
  {
    const int r = tid >> 1, hf = tid & 1;
    const int g = wg0 + r;
    __bf16* dst = A + r * 192 + hf * 32;
    if (g < Nn) {
      const int bb = batch[g];
      const float sc = dinv[g];
      stage32(dst + 0, x + (size_t)g * H + hf * 32);
      stage32_scaled(dst + 64, agg + (size_t)g * H + hf * 32, sc);
      stage32(dst + 128, u + bb * H + hf * 32);
    } else {
      uint4 z = {0, 0, 0, 0};
#pragma unroll
      for (int sg = 0; sg < 3; ++sg)
#pragma unroll
        for (int j = 0; j < 4; ++j) ((uint4*)(dst + sg * 64))[j] = z;
    }
  }
  const int wave = tid >> 5, lane = tid & 31, l15 = lane & 15, hi = lane >> 4;
  f32x8 acc[4] = {};
  wave_gemm<6, 192, 192>(A + wave * 16 * 192, W1t, acc, l15, hi);
  __bf16* Hw = Hd + wave * 16 * H;
  store_hidden_relu(Hw, acc, b1, l15, hi);
  f32x8 acc2[4] = {};
  wave_gemm<2, H, H>(Hw, W2t, acc2, l15, hi);
  const int r0 = wg0 + wave * 16 + hi * 8;
#pragma unroll
  for (int t = 0; t < 4; ++t) {
    float bn = b2[t * 16 + l15];
#pragma unroll
    for (int r = 0; r < 8; ++r) {
      int g = r0 + r;
      if (g < Nn) {
        int c = t * 16 + l15;
        xout[(size_t)g * H + c] = acc2[t][r] + bn + x[(size_t)g * H + c];
      }
    }
  }
}

// Decoder: WMMA 64->64 relu, then tiny f32 64->3.
__global__ void __launch_bounds__(128) k_decode(
    const float* __restrict__ x,
    const __bf16* __restrict__ W1t, const float* __restrict__ b1,
    const float* __restrict__ w2, const float* __restrict__ b2,
    float* __restrict__ out, int Nn) {
  __shared__ __attribute__((aligned(16))) __bf16 A[64 * H];
  __shared__ __attribute__((aligned(16))) __bf16 Hd[64 * H];
  const int wg0 = blockIdx.x * 64;
  const int tid = threadIdx.x;
  {
    const int r = tid >> 1, hf = tid & 1;
    const int g = wg0 + r;
    __bf16* dst = A + r * H + hf * 32;
    if (g < Nn) stage32(dst, x + (size_t)g * H + hf * 32);
    else {
      uint4 z = {0, 0, 0, 0};
#pragma unroll
      for (int j = 0; j < 4; ++j) ((uint4*)dst)[j] = z;
    }
  }
  const int wave = tid >> 5, lane = tid & 31, l15 = lane & 15, hi = lane >> 4;
  f32x8 acc[4] = {};
  wave_gemm<2, H, H>(A + wave * 16 * H, W1t, acc, l15, hi);
  __bf16* Hw = Hd + wave * 16 * H;
  store_hidden_relu(Hw, acc, b1, l15, hi);
  for (int task = lane; task < 16 * NOUT; task += 32) {
    int r = task / NOUT, c = task % NOUT;
    int g = wg0 + wave * 16 + r;
    if (g < Nn) {
      float s = b2[c];
      for (int k = 0; k < H; ++k) s += (float)Hw[r * H + k] * w2[k * NOUT + c];
      out[(size_t)g * NOUT + c] = s;
    }
  }
}

// ================================ host side =================================
extern "C" void kernel_launch(void* const* d_in, const int* in_sizes, int n_in,
                              void* d_out, int out_size, void* d_ws, size_t ws_size,
                              hipStream_t stream) {
  (void)n_in; (void)out_size; (void)ws_size;  // ws needs ~145 MB
  const int N = in_sizes[0] / NODE_IN;
  const int E = in_sizes[2] / EDGE_IN;

  const float* x_in  = (const float*)d_in[0];
  const int*   ei    = (const int*)d_in[1];   // (2,E): row=ei[0:E], col=ei[E:2E]
  const float* eattr = (const float*)d_in[2];
  const float* cond  = (const float*)d_in[3];
  const int*   batch = (const int*)d_in[4];

  // params leaves in jax pytree order (sorted dict keys):
  // cond_enc, dec, edge_enc, layers[0..2]{edge,node1,node2}, node_enc;
  // each mlp = (l1,l2), each lin = (b,w).
  const float* ce_b1 = (const float*)d_in[5];
  const float* ce_w1 = (const float*)d_in[6];
  const float* ce_b2 = (const float*)d_in[7];
  const float* ce_w2 = (const float*)d_in[8];
  const float* de_b1 = (const float*)d_in[9];
  const float* de_w1 = (const float*)d_in[10];
  const float* de_b2 = (const float*)d_in[11];
  const float* de_w2 = (const float*)d_in[12];
  const float* ee_b1 = (const float*)d_in[13];
  const float* ee_w1 = (const float*)d_in[14];
  const float* ee_b2 = (const float*)d_in[15];
  const float* ee_w2 = (const float*)d_in[16];
  const float *L_eb1[3], *L_ew1[3], *L_eb2[3], *L_ew2[3];
  const float *L_n1b1[3], *L_n1w1[3], *L_n1b2[3], *L_n1w2[3];
  const float *L_n2b1[3], *L_n2w1[3], *L_n2b2[3], *L_n2w2[3];
  for (int l = 0; l < NLAYERS; ++l) {
    int b = 17 + 12 * l;
    L_eb1[l]  = (const float*)d_in[b + 0];  L_ew1[l]  = (const float*)d_in[b + 1];
    L_eb2[l]  = (const float*)d_in[b + 2];  L_ew2[l]  = (const float*)d_in[b + 3];
    L_n1b1[l] = (const float*)d_in[b + 4];  L_n1w1[l] = (const float*)d_in[b + 5];
    L_n1b2[l] = (const float*)d_in[b + 6];  L_n1w2[l] = (const float*)d_in[b + 7];
    L_n2b1[l] = (const float*)d_in[b + 8];  L_n2w1[l] = (const float*)d_in[b + 9];
    L_n2b2[l] = (const float*)d_in[b + 10]; L_n2w2[l] = (const float*)d_in[b + 11];
  }
  const float* ne_b1 = (const float*)d_in[53];
  const float* ne_w1 = (const float*)d_in[54];
  const float* ne_b2 = (const float*)d_in[55];
  const float* ne_w2 = (const float*)d_in[56];

  // ---- workspace carve-up ----
  char* ws = (char*)d_ws;
  size_t off = 0;
  auto carve = [&](size_t bytes) -> char* {
    char* p = ws + off;
    off += (bytes + 255) & ~(size_t)255;
    return p;
  };
  float* U   = (float*)carve((size_t)NB * H * 4);
  float* DEG = (float*)carve((size_t)N * 4);
  float* XA  = (float*)carve((size_t)N * H * 4);
  float* XB  = (float*)carve((size_t)N * H * 4);
  float* AGG = (float*)carve((size_t)N * H * 4);
  float* EA  = (float*)carve((size_t)E * H * 4);
  __bf16* wNE1 = (__bf16*)carve(64 * 32 * 2);
  __bf16* wNE2 = (__bf16*)carve(64 * 64 * 2);
  __bf16* wEE1 = (__bf16*)carve(64 * 32 * 2);
  __bf16* wEE2 = (__bf16*)carve(64 * 64 * 2);
  __bf16* wDE1 = (__bf16*)carve(64 * 64 * 2);
  __bf16 *wE1[3], *wE2[3], *wN11[3], *wN12[3], *wN21[3], *wN22[3];
  for (int l = 0; l < NLAYERS; ++l) {
    wE1[l]  = (__bf16*)carve(64 * 256 * 2);
    wE2[l]  = (__bf16*)carve(64 * 64 * 2);
    wN11[l] = (__bf16*)carve(64 * 128 * 2);
    wN12[l] = (__bf16*)carve(64 * 64 * 2);
    wN21[l] = (__bf16*)carve(64 * 192 * 2);
    wN22[l] = (__bf16*)carve(64 * 64 * 2);
  }

  auto pack = [&](const float* w, __bf16* dst, int din, int kp) {
    int total = 64 * kp;
    k_pack_w<<<(total + 255) / 256, 256, 0, stream>>>(w, dst, din, kp);
  };
  pack(ne_w1, wNE1, NODE_IN, 32);
  pack(ne_w2, wNE2, 64, 64);
  pack(ee_w1, wEE1, EDGE_IN, 32);
  pack(ee_w2, wEE2, 64, 64);
  pack(de_w1, wDE1, 64, 64);
  for (int l = 0; l < NLAYERS; ++l) {
    pack(L_ew1[l],  wE1[l],  256, 256);
    pack(L_ew2[l],  wE2[l],  64, 64);
    pack(L_n1w1[l], wN11[l], 128, 128);
    pack(L_n1w2[l], wN12[l], 64, 64);
    pack(L_n2w1[l], wN21[l], 192, 192);
    pack(L_n2w2[l], wN22[l], 64, 64);
  }

  k_cond_enc<<<1, 64, 0, stream>>>(cond, ce_w1, ce_b1, ce_w2, ce_b2, U);

  const int nwgN = (N + 63) / 64;
  const int nwgE = (E + 63) / 64;
  k_node_enc<<<nwgN, 128, 0, stream>>>(x_in, wNE1, ne_b1, wNE2, ne_b2, XA, N);
  k_edge_enc<<<nwgE, 128, 0, stream>>>(eattr, wEE1, ee_b1, wEE2, ee_b2, EA, E);

  k_zero<<<512, 256, 0, stream>>>(DEG, N);
  k_deg_count<<<(E + 255) / 256, 256, 0, stream>>>(ei + E, DEG, E);
  k_deg_inv<<<(N + 255) / 256, 256, 0, stream>>>(DEG, N);

  const float* xc = XA;
  float* xn = XB;
  for (int l = 0; l < NLAYERS; ++l) {
    k_zero<<<2048, 256, 0, stream>>>(AGG, N * H);
    k_edge_layer<<<nwgE, 128, 0, stream>>>(
        xc, ei, U, batch, EA,
        wE1[l], L_eb1[l], wE2[l], L_eb2[l],
        wN11[l], L_n1b1[l], wN12[l], L_n1b2[l], AGG, E);
    k_node_layer<<<nwgN, 128, 0, stream>>>(
        xc, AGG, DEG, U, batch,
        wN21[l], L_n2b1[l], wN22[l], L_n2b2[l], xn, N);
    float* tmp = (float*)xc; xc = xn; xn = tmp;
  }
  k_decode<<<nwgN, 128, 0, stream>>>(xc, wDE1, de_b1, de_w2, de_b2,
                                     (float*)d_out, N);
}
